// FSCTNet_13237089206895
// MI455X (gfx1250) — matrix operand order repaired and checked by reading, not compiled, not run on hardware
//
#include <hip/hip_runtime.h>
#include <hip/hip_bf16.h>
#include <cstdint>
#include <cstddef>

// ---------------------------------------------------------------------------
// Types for CDNA5 WMMA (wave32): v_wmma_f32_16x16x32_bf16
// ---------------------------------------------------------------------------
typedef __attribute__((ext_vector_type(16))) __bf16 v16bf;
typedef __attribute__((ext_vector_type(8)))  float  v8f;

#define EPS_BN 1e-5f

// ---------------------------------------------------------------------------
// GEMM: C[Mp,Np] = relu?(A[Mp,lda] @ W + bias)
//   A   : f32, lda multiple of 32 (rows clamped, never OOB)
//   Wt  : bf16, transposed weights [Np, Ka] (Ka mult of 32, Np mult of 128),
//         zero padded beyond (Kdim, Nreal)
//   One wave -> 32x64 tile via 2x4 v_wmma_f32_16x16x32_bf16 fragments
//   (8 WMMAs / K-step, 2 loads per WMMA). Block = 8 waves -> 128x128 tile.
// ---------------------------------------------------------------------------
static __device__ __forceinline__ v16bf load_a_frag(const float* ap, int k0) {
    const float4* p = (const float4*)(ap + k0);        // K = half*16 + 0..7
    float4 x0 = p[0], x1 = p[1];
    const float4* q = (const float4*)(ap + k0 + 8);    // K = half*16 + 8..15
    float4 x2 = q[0], x3 = q[1];
    v16bf a;
    a[0]  = (__bf16)x0.x; a[1]  = (__bf16)x0.y; a[2]  = (__bf16)x0.z; a[3]  = (__bf16)x0.w;
    a[4]  = (__bf16)x1.x; a[5]  = (__bf16)x1.y; a[6]  = (__bf16)x1.z; a[7]  = (__bf16)x1.w;
    a[8]  = (__bf16)x2.x; a[9]  = (__bf16)x2.y; a[10] = (__bf16)x2.z; a[11] = (__bf16)x2.w;
    a[12] = (__bf16)x3.x; a[13] = (__bf16)x3.y; a[14] = (__bf16)x3.z; a[15] = (__bf16)x3.w;
    return a;
}

__global__ __launch_bounds__(256) void k_gemm(
    const float* __restrict__ A, int lda, int Mp,
    const __bf16* __restrict__ Wt, int Ka,
    const float* __restrict__ bias, int Nreal, int doRelu,
    float* __restrict__ C, int ldc)
{
    const int lane = threadIdx.x & 31;
    const int wave = threadIdx.x >> 5;
    const int wy = wave >> 1, wx = wave & 1;
    const int rowBase = blockIdx.y * 128 + wy * 32;
    const int colBase = blockIdx.x * 128 + wx * 64;
    const int m = lane & 15, half = lane >> 4;

    v8f acc[2][4];
#pragma unroll
    for (int i = 0; i < 2; ++i)
#pragma unroll
        for (int j = 0; j < 4; ++j) acc[i][j] = (v8f){};

    const int r0 = rowBase + m;
    const int r1 = rowBase + 16 + m;
    // clamp to last valid row: loads become unconditional, stores stay guarded
    const int rc0 = r0 < Mp ? r0 : Mp - 1;
    const int rc1 = r1 < Mp ? r1 : Mp - 1;
    const float* ap0 = A + (size_t)rc0 * lda + half * 16;
    const float* ap1 = A + (size_t)rc1 * lda + half * 16;
    const __bf16* wp[4];
#pragma unroll
    for (int j = 0; j < 4; ++j)
        wp[j] = Wt + (size_t)(colBase + j * 16 + m) * Ka + half * 16;

    for (int k0 = 0; k0 < Ka; k0 += 32) {
        // speculative prefetch of next K-slab (dropped silently at buffer end)
        __builtin_prefetch(ap0 + k0 + 32, 0, 1);
        __builtin_prefetch(ap1 + k0 + 32, 0, 1);
        __builtin_prefetch(wp[0] + k0 + 32, 0, 1);
        __builtin_prefetch(wp[3] + k0 + 32, 0, 1);
        v16bf a0 = load_a_frag(ap0, k0);
        v16bf a1 = load_a_frag(ap1, k0);
        v16bf b[4];
#pragma unroll
        for (int j = 0; j < 4; ++j) b[j] = *(const v16bf*)(wp[j] + k0);
#pragma unroll
        for (int j = 0; j < 4; ++j) {
            acc[0][j] = __builtin_amdgcn_wmma_f32_16x16x32_bf16(false, a0, false, b[j], (short)0, acc[0][j], false, false);
            acc[1][j] = __builtin_amdgcn_wmma_f32_16x16x32_bf16(false, a1, false, b[j], (short)0, acc[1][j], false, false);
        }
    }

    // D layout: lane -> col = colBase + fj*16 + (lane&15); elem v -> row = fr*16 + half*8 + v
#pragma unroll
    for (int fr = 0; fr < 2; ++fr) {
#pragma unroll
        for (int fc = 0; fc < 4; ++fc) {
            int col = colBase + fc * 16 + m;
            float bv = (col < Nreal) ? bias[col] : 0.f;
#pragma unroll
            for (int v = 0; v < 8; ++v) {
                int row = rowBase + fr * 16 + half * 8 + v;
                if (row < Mp) {
                    float x = acc[fr][fc][v] + bv;
                    if (doRelu) x = fmaxf(x, 0.f);
                    C[(size_t)row * ldc + col] = x;
                }
            }
        }
    }
}

// ---------------------------------------------------------------------------
// Weight transpose + f32 -> bf16 convert with zero padding:
//   W[Kdim, Nreal] (row major)  ->  Wt[Np, Ka]
// ---------------------------------------------------------------------------
__global__ void k_cvtw(const float* __restrict__ W, int Kdim, int Nreal,
                       int Ka, long long total, __bf16* __restrict__ Wt)
{
    long long t = (long long)blockIdx.x * blockDim.x + threadIdx.x;
    if (t >= total) return;
    int kk = (int)(t % Ka);
    long long nn = t / Ka;
    float v = (nn < Nreal && kk < Kdim) ? W[(size_t)kk * Nreal + nn] : 0.f;
    Wt[t] = (__bf16)v;
}

// ---------------------------------------------------------------------------
// Deterministic farthest point sampling (single block, LDS argmax reduce).
// Tie-break: lowest index (matches jnp.argmax).
// ---------------------------------------------------------------------------
__global__ __launch_bounds__(1024) void k_fps(const float* __restrict__ pos,
                                              int Npts, int n, int* __restrict__ idxOut)
{
    __shared__ float sv[1024];
    __shared__ int   si[1024];
    __shared__ float spt[3];
    const int tid = threadIdx.x;

    float d[16], px[16], py[16], pz[16];
    const float ox = pos[0], oy = pos[1], oz = pos[2];
#pragma unroll
    for (int j = 0; j < 16; ++j) {
        int i = tid + j * 1024;
        if (i < Npts) {
            px[j] = pos[i * 3]; py[j] = pos[i * 3 + 1]; pz[j] = pos[i * 3 + 2];
            float dx = px[j] - ox, dy = py[j] - oy, dz = pz[j] - oz;
            d[j] = dx * dx + dy * dy + dz * dz;
        } else { d[j] = -1.f; px[j] = py[j] = pz[j] = 0.f; }
    }
    if (tid == 0) idxOut[0] = 0;

    for (int it = 1; it < n; ++it) {
        float bv = -1e30f; int bi = 0x7fffffff;
#pragma unroll
        for (int j = 0; j < 16; ++j) {
            int i = tid + j * 1024;
            if (i < Npts && d[j] > bv) { bv = d[j]; bi = i; }
        }
        sv[tid] = bv; si[tid] = bi;
        __syncthreads();
        for (int s = 512; s > 0; s >>= 1) {
            if (tid < s) {
                float va = sv[tid], vb = sv[tid + s];
                int ia = si[tid], ib = si[tid + s];
                if (vb > va || (vb == va && ib < ia)) { sv[tid] = vb; si[tid] = ib; }
            }
            __syncthreads();
        }
        if (tid == 0) {
            int b = si[0];
            idxOut[it] = b;
            spt[0] = pos[b * 3]; spt[1] = pos[b * 3 + 1]; spt[2] = pos[b * 3 + 2];
        }
        __syncthreads();
        float cx = spt[0], cy = spt[1], cz = spt[2];
#pragma unroll
        for (int j = 0; j < 16; ++j) {
            float dx = px[j] - cx, dy = py[j] - cy, dz = pz[j] - cz;
            d[j] = fminf(d[j], dx * dx + dy * dy + dz * dz);
        }
        __syncthreads();
    }
}

// gather 3-vectors: out[i] = src[idx[i]]
__global__ void k_gather3(const float* __restrict__ src, const int* __restrict__ idx,
                          int n, float* __restrict__ out)
{
    int t = blockIdx.x * blockDim.x + threadIdx.x;
    if (t >= n * 3) return;
    int r = t / 3, c = t % 3;
    out[t] = src[idx[r] * 3 + c];
}

// ---------------------------------------------------------------------------
// K=64 nearest neighbors per centroid (block per centroid).
// Packed key (d2_bits<<32)|idx gives exact ascending order w/ index tiebreak.
// ---------------------------------------------------------------------------
__global__ __launch_bounds__(256) void k_knn_topk(
    const float* __restrict__ cpos, const float* __restrict__ pos, int Npts,
    float r2, int* __restrict__ nidx, float* __restrict__ mask)
{
    __shared__ unsigned long long sk[256];
    const int b = blockIdx.x, tid = threadIdx.x;
    const float cx = cpos[b * 3], cy = cpos[b * 3 + 1], cz = cpos[b * 3 + 2];
    unsigned long long prev = 0ull;
    for (int j = 0; j < 64; ++j) {
        unsigned long long best = ~0ull;
        for (int i = tid; i < Npts; i += 256) {
            float dx = pos[i * 3] - cx, dy = pos[i * 3 + 1] - cy, dz = pos[i * 3 + 2] - cz;
            float d2 = dx * dx + dy * dy + dz * dz;
            unsigned long long key = ((unsigned long long)__float_as_uint(d2) << 32) | (unsigned)i;
            if (key >= prev && key < best) best = key;
        }
        sk[tid] = best;
        __syncthreads();
        for (int s = 128; s > 0; s >>= 1) {
            if (tid < s && sk[tid + s] < sk[tid]) sk[tid] = sk[tid + s];
            __syncthreads();
        }
        unsigned long long kmin = sk[0];
        if (tid == 0) {
            int sel = (int)(kmin & 0xffffffffull);
            float d2v = __uint_as_float((unsigned)(kmin >> 32));
            nidx[b * 64 + j] = sel;
            mask[b * 64 + j] = (d2v <= r2) ? 1.f : 0.f;
        }
        prev = kmin + 1ull;
        __syncthreads();
    }
}

// scalar sum (single block)
__global__ __launch_bounds__(1024) void k_sum(const float* __restrict__ v, int n,
                                              float* __restrict__ out)
{
    __shared__ float s[1024];
    float a = 0.f;
    for (int i = threadIdx.x; i < n; i += 1024) a += v[i];
    s[threadIdx.x] = a;
    __syncthreads();
    for (int st = 512; st > 0; st >>= 1) {
        if (threadIdx.x < st) s[threadIdx.x] += s[threadIdx.x + st];
        __syncthreads();
    }
    if (threadIdx.x == 0) *out = s[0];
}

// Build SA input: A[(ci*64+k), :] = [feat[nidx], pos[nidx]-cpos] zero-padded to Cpad
__global__ void k_build_sa(const float* __restrict__ feat, int featDim,
                           const float* __restrict__ pos, const float* __restrict__ cpos,
                           const int* __restrict__ nidx, int n, int Cpad,
                           float* __restrict__ A)
{
    long long t = (long long)blockIdx.x * blockDim.x + threadIdx.x;
    long long total = (long long)n * 64 * Cpad;
    if (t >= total) return;
    int c = (int)(t % Cpad);
    long long r = t / Cpad;
    int k = (int)(r % 64);
    int ci = (int)(r / 64);
    int src = nidx[ci * 64 + k];
    float v = 0.f;
    if (c < featDim) v = feat[(size_t)src * featDim + c];
    else if (c < featDim + 3) {
        int dd = c - featDim;
        v = pos[src * 3 + dd] - cpos[ci * 3 + dd];
    }
    A[t] = v;
}

// ---------------------------------------------------------------------------
// BN statistics, two-pass, coalesced, deterministic (no atomics).
// Pass 1: blocks = (Nc/32, 64 row-chunks); lanes map to consecutive channels.
// Pass 2: per-channel reduction of the 64 partials.
// ---------------------------------------------------------------------------
__global__ __launch_bounds__(256) void k_bnstats_part(
    const float* __restrict__ H, int Mreal, int ldh,
    const float* __restrict__ mask,
    float* __restrict__ sPart, float* __restrict__ qPart)
{
    __shared__ float ss[8][33], sq[8][33];
    const int tx = threadIdx.x & 31;
    const int ty = threadIdx.x >> 5;
    const int c = blockIdx.x * 32 + tx;
    const int chunk = blockIdx.y;           // 0..63
    float s = 0.f, q = 0.f;
    for (int r = chunk * 8 + ty; r < Mreal; r += 512) {
        float w = mask ? mask[r] : 1.f;
        float h = H[(size_t)r * ldh + c];
        s += h * w; q += h * h * w;
    }
    ss[ty][tx] = s; sq[ty][tx] = q;
    __syncthreads();
    if (ty == 0) {
#pragma unroll
        for (int j = 1; j < 8; ++j) { s += ss[j][tx]; q += sq[j][tx]; }
        sPart[(size_t)chunk * ldh + c] = s;
        qPart[(size_t)chunk * ldh + c] = q;
    }
}

__global__ void k_bnstats_fin(const float* __restrict__ sPart,
                              const float* __restrict__ qPart, int Nc,
                              const float* __restrict__ cntp, int Mreal,
                              float* __restrict__ mean, float* __restrict__ var)
{
    int c = blockIdx.x * blockDim.x + threadIdx.x;
    if (c >= Nc) return;
    float s = 0.f, q = 0.f;
    for (int j = 0; j < 64; ++j) {
        s += sPart[(size_t)j * Nc + c];
        q += qPart[(size_t)j * Nc + c];
    }
    float cnt = cntp ? *cntp : (float)Mreal;
    float mu = s / cnt;
    float vv = q / cnt - mu * mu;
    mean[c] = mu;
    var[c] = vv > 0.f ? vv : 0.f;
}

// In-place BN apply (ldh == Nreal for all BN layers), optional trailing relu.
__global__ void k_bnapply(float* __restrict__ H, long long total, int ldh,
                          const float* __restrict__ mean, const float* __restrict__ var,
                          const float* __restrict__ g, const float* __restrict__ be,
                          int reluAfter)
{
    long long t = (long long)blockIdx.x * blockDim.x + threadIdx.x;
    if (t >= total) return;
    int c = (int)(t % ldh);
    float v = H[t];
    v = (v - mean[c]) * rsqrtf(var[c] + EPS_BN) * g[c] + be[c];
    if (reluAfter) v = fmaxf(v, 0.f);
    H[t] = v;
}

// masked max over groups of Kg rows
__global__ void k_maxpool(const float* __restrict__ H, int ldh,
                          const float* __restrict__ mask,
                          int nG, int Kg, int C, float* __restrict__ out)
{
    long long t = (long long)blockIdx.x * blockDim.x + threadIdx.x;
    long long total = (long long)nG * C;
    if (t >= total) return;
    int c = (int)(t % C);
    int gi = (int)(t / C);
    float m = -3.4e38f;
    for (int k = 0; k < Kg; ++k) {
        long long r = (long long)gi * Kg + k;
        if (!mask || mask[r] > 0.f) m = fmaxf(m, H[r * (size_t)ldh + c]);
    }
    out[t] = m;
}

// k-NN (k<=3) per target over small source set, 1/d2 weights
__global__ void k_knn_small(const float* __restrict__ tpos, int Nt,
                            const float* __restrict__ spos, int Ns, int k,
                            int* __restrict__ idx, float* __restrict__ w)
{
    int t = blockIdx.x * blockDim.x + threadIdx.x;
    if (t >= Nt) return;
    float tx = tpos[t * 3], ty = tpos[t * 3 + 1], tz = tpos[t * 3 + 2];
    float bd0 = 3.4e38f, bd1 = 3.4e38f, bd2 = 3.4e38f;
    int bi0 = 0, bi1 = 0, bi2 = 0;
    for (int s = 0; s < Ns; ++s) {
        float dx = spos[s * 3] - tx, dy = spos[s * 3 + 1] - ty, dz = spos[s * 3 + 2] - tz;
        float d2 = dx * dx + dy * dy + dz * dz;
        if (d2 < bd0)      { bd2 = bd1; bi2 = bi1; bd1 = bd0; bi1 = bi0; bd0 = d2; bi0 = s; }
        else if (d2 < bd1) { bd2 = bd1; bi2 = bi1; bd1 = d2; bi1 = s; }
        else if (d2 < bd2) { bd2 = d2; bi2 = s; }
    }
    float bds[3] = {bd0, bd1, bd2};
    int   bis[3] = {bi0, bi1, bi2};
    for (int j = 0; j < k; ++j) {
        idx[t * k + j] = bis[j];
        w[t * k + j] = 1.f / fmaxf(bds[j], 1e-16f);
    }
}

// inverse-distance weighted gather: out[t,c] = sum_j w_j * X[idx_j, c] / sum_j w_j
__global__ void k_interp(const float* __restrict__ X, int ldx,
                         const int* __restrict__ idx, const float* __restrict__ w, int k,
                         int Nt, int C, float* __restrict__ out, int ldo)
{
    long long t = (long long)blockIdx.x * blockDim.x + threadIdx.x;
    long long total = (long long)Nt * C;
    if (t >= total) return;
    int c = (int)(t % C);
    long long r = t / C;
    float acc = 0.f, ws = 0.f;
    for (int j = 0; j < k; ++j) {
        float wj = w[r * k + j];
        acc += wj * X[(size_t)idx[r * k + j] * ldx + c];
        ws += wj;
    }
    out[(size_t)r * ldo + c] = acc / ws;
}

// 2D copy into a (possibly wider) destination at column offset
__global__ void k_copy2d(const float* __restrict__ src, int lds, int Nt, int C,
                         float* __restrict__ dst, int ldo, int colOff)
{
    long long t = (long long)blockIdx.x * blockDim.x + threadIdx.x;
    long long total = (long long)Nt * C;
    if (t >= total) return;
    int c = (int)(t % C);
    long long r = t / C;
    dst[(size_t)r * ldo + colOff + c] = src[(size_t)r * lds + c];
}

__global__ void k_fill(float* __restrict__ p, long long n)
{
    long long t = (long long)blockIdx.x * blockDim.x + threadIdx.x;
    if (t < n) p[t] = 0.f;
}

// log_softmax over 13 classes, transposed output [C, N0]
__global__ void k_lsm(const float* __restrict__ lg, int ldl, int Npts,
                      float* __restrict__ out)
{
    int i = blockIdx.x * blockDim.x + threadIdx.x;
    if (i >= Npts) return;
    float v[13];
    float m = -3.4e38f;
#pragma unroll
    for (int c = 0; c < 13; ++c) { v[c] = lg[(size_t)i * ldl + c]; m = fmaxf(m, v[c]); }
    float s = 0.f;
#pragma unroll
    for (int c = 0; c < 13; ++c) s += expf(v[c] - m);
    float l = logf(s);
#pragma unroll
    for (int c = 0; c < 13; ++c) out[(size_t)c * Npts + i] = v[c] - m - l;
}

// ---------------------------------------------------------------------------
// Host orchestration
// ---------------------------------------------------------------------------
static inline unsigned g1d(long long total, int bs = 256) {
    return (unsigned)((total + bs - 1) / bs);
}

extern "C" void kernel_launch(void* const* d_in, const int* in_sizes, int n_in,
                              void* d_out, int out_size, void* d_ws, size_t ws_size,
                              hipStream_t stream)
{
    (void)in_sizes; (void)n_in; (void)out_size; (void)ws_size;

    const float* pos = (const float*)d_in[0];
    const float* xin = (const float*)d_in[1];
    auto P = [&](int i) { return (const float*)d_in[i]; };

    // bump allocator over workspace
    char* ws = (char*)d_ws;
    size_t off = 0;
    auto alloc = [&](size_t bytes) -> char* {
        char* p = ws + off;
        off = (off + bytes + 255) & ~(size_t)255;
        return p;
    };
    auto allocf = [&](size_t n) { return (float*)alloc(n * 4); };
    auto alloci = [&](size_t n) { return (int*)alloc(n * 4); };
    auto allocb = [&](size_t n) { return (__bf16*)alloc(n * 2); };

    // ---- bf16 transposed weights (Np multiple of 128 for 128-wide tiles) ---
    struct WDesc { int widx; int Kdim; int N; int Ka; int Np; __bf16* wt; };
    WDesc wd[17] = {
        {2,    6,  128,   32,  128, nullptr},   // sa1.0
        {6,  128,  256,  128,  256, nullptr},   // sa1.1
        {10, 256,  512,  256,  512, nullptr},   // sa1.2
        {14, 515,  512,  544,  512, nullptr},   // sa2.0
        {18, 512, 1024,  512, 1024, nullptr},   // sa2.1
        {22,1024, 1024, 1024, 1024, nullptr},   // sa2.2
        {26,1027, 1024, 1056, 1024, nullptr},   // sa3.0
        {30,1024, 2048, 1024, 2048, nullptr},   // sa3.1
        {34,2048, 2048, 2048, 2048, nullptr},   // sa3.2
        {38,3072, 1024, 3072, 1024, nullptr},   // fp3.0
        {42,1024, 1024, 1024, 1024, nullptr},   // fp3.1
        {46,1536, 1024, 1536, 1024, nullptr},   // fp2.0
        {50,1024, 1024, 1024, 1024, nullptr},   // fp2.1
        {54,1027, 1024, 1056, 1024, nullptr},   // fp1.0
        {58,1024, 1024, 1024, 1024, nullptr},   // fp1.1
        {62,1024, 1024, 1024, 1024, nullptr},   // head W1
        {66,1024,   13, 1024,  128, nullptr},   // head W2 (N padded to 128)
    };
    for (int i = 0; i < 17; ++i)
        wd[i].wt = allocb((size_t)wd[i].Np * wd[i].Ka);
    for (int i = 0; i < 17; ++i) {
        long long tot = (long long)wd[i].Np * wd[i].Ka;
        k_cvtw<<<g1d(tot), 256, 0, stream>>>(P(wd[i].widx), wd[i].Kdim, wd[i].N,
                                             wd[i].Ka, tot, wd[i].wt);
    }

    // ---- persistent buffers ------------------------------------------------
    int*   idx1  = alloci(1638);
    float* cpos1 = allocf(1638 * 3);
    int*   nidx1 = alloci(1638 * 64);
    float* mask1 = allocf(1638 * 64);
    float* x1    = allocf((size_t)1638 * 512);
    int*   idx2  = alloci(82);
    float* cpos2 = allocf(82 * 3);
    int*   nidx2 = alloci(82 * 64);
    float* mask2 = allocf(82 * 64);
    float* x2    = allocf((size_t)82 * 1024);
    float* x3    = allocf(2048);
    float* zero3 = allocf(3);
    int*   iI3   = alloci(82);        float* wI3 = allocf(82);
    int*   iI2   = alloci(1638 * 3);  float* wI2 = allocf(1638 * 3);
    int*   iI1   = alloci(16384 * 3); float* wI1 = allocf(16384 * 3);
    float* cnt1  = allocf(1);
    float* cnt2  = allocf(1);
    float* meanB = allocf(2048);
    float* varB  = allocf(2048);
    float* sPart = allocf((size_t)64 * 2048);
    float* qPart = allocf((size_t)64 * 2048);
    float* lg    = allocf((size_t)16384 * 128);
    float* slotA = allocf((size_t)104832 * 256);   // 107 MB arena
    float* slotB = allocf((size_t)104832 * 512);   // 215 MB arena

    // ---- common helpers ----------------------------------------------------
    auto gemm = [&](const float* A, int lda, int Mp, const WDesc& w,
                    const float* bias, int relu, float* C) {
        dim3 g(w.Np / 128, (unsigned)((Mp + 127) / 128));
        k_gemm<<<g, 256, 0, stream>>>(A, lda, Mp, w.wt, w.Ka, bias, w.N, relu, C, w.Np);
    };
    auto bn = [&](float* H, int Mreal, int Mp, int Nc, const float* mask,
                  const float* cnt, const float* gg, const float* bb, int reluAfter) {
        dim3 gs((unsigned)(Nc / 32), 64);
        k_bnstats_part<<<gs, 256, 0, stream>>>(H, Mreal, Nc, mask, sPart, qPart);
        k_bnstats_fin<<<g1d(Nc), 256, 0, stream>>>(sPart, qPart, Nc, cnt, Mreal, meanB, varB);
        long long tot = (long long)Mp * Nc;
        k_bnapply<<<g1d(tot), 256, 0, stream>>>(H, tot, Nc, meanB, varB, gg, bb, reluAfter);
    };
    auto mlpLayer = [&](int wi, const float* A, int lda, int Mreal, int Mp,
                        const float* mask, const float* cnt, float* C) {
        gemm(A, lda, Mp, wd[wi], P(wd[wi].widx + 1), 1, C);
        bn(C, Mreal, Mp, wd[wi].N, mask, cnt, P(wd[wi].widx + 2), P(wd[wi].widx + 3), 0);
    };

    // ======================= SA1 (N0=16384 -> N1=1638) ======================
    k_fps<<<1, 1024, 0, stream>>>(pos, 16384, 1638, idx1);
    k_gather3<<<g1d(1638 * 3), 256, 0, stream>>>(pos, idx1, 1638, cpos1);
    k_knn_topk<<<1638, 256, 0, stream>>>(cpos1, pos, 16384, 0.2f * 0.2f, nidx1, mask1);
    k_sum<<<1, 1024, 0, stream>>>(mask1, 1638 * 64, cnt1);
    {
        long long tot = 104832ll * 32;
        k_build_sa<<<g1d(tot), 256, 0, stream>>>(xin, 3, pos, cpos1, nidx1, 1638, 32, slotA);
    }
    mlpLayer(0, slotA,  32, 104832, 104832, mask1, cnt1, slotB);   // -> [*,128]
    mlpLayer(1, slotB, 128, 104832, 104832, mask1, cnt1, slotA);   // -> [*,256]
    mlpLayer(2, slotA, 256, 104832, 104832, mask1, cnt1, slotB);   // -> [*,512]
    k_maxpool<<<g1d(1638ll * 512), 256, 0, stream>>>(slotB, 512, mask1, 1638, 64, 512, x1);

    // ======================= SA2 (1638 -> 82) ===============================
    k_fps<<<1, 1024, 0, stream>>>(cpos1, 1638, 82, idx2);
    k_gather3<<<g1d(82 * 3), 256, 0, stream>>>(cpos1, idx2, 82, cpos2);
    k_knn_topk<<<82, 256, 0, stream>>>(cpos2, cpos1, 1638, 0.4f * 0.4f, nidx2, mask2);
    k_sum<<<1, 1024, 0, stream>>>(mask2, 82 * 64, cnt2);
    {
        long long tot = 5248ll * 544;
        k_build_sa<<<g1d(tot), 256, 0, stream>>>(x1, 512, cpos1, cpos2, nidx2, 82, 544, slotA);
    }
    mlpLayer(3, slotA,  544, 5248, 5248, mask2, cnt2, slotB);      // -> [*,512]
    mlpLayer(4, slotB,  512, 5248, 5248, mask2, cnt2, slotA);      // -> [*,1024]
    mlpLayer(5, slotA, 1024, 5248, 5248, mask2, cnt2, slotB);      // -> [*,1024]
    k_maxpool<<<g1d(82ll * 1024), 256, 0, stream>>>(slotB, 1024, mask2, 82, 64, 1024, x2);

    // ======================= SA3 (global, 82 rows) ==========================
    k_fill<<<g1d(96ll * 1056), 256, 0, stream>>>(slotA, 96ll * 1056);
    k_copy2d<<<g1d(82ll * 1024), 256, 0, stream>>>(x2, 1024, 82, 1024, slotA, 1056, 0);
    k_copy2d<<<g1d(82ll * 3), 256, 0, stream>>>(cpos2, 3, 82, 3, slotA, 1056, 1024);
    mlpLayer(6, slotA, 1056, 82, 96, nullptr, nullptr, slotB);     // -> [*,1024]
    mlpLayer(7, slotB, 1024, 82, 96, nullptr, nullptr, slotA);     // -> [*,2048]
    mlpLayer(8, slotA, 2048, 82, 96, nullptr, nullptr, slotB);     // -> [*,2048]
    k_maxpool<<<g1d(2048), 256, 0, stream>>>(slotB, 2048, nullptr, 1, 82, 2048, x3);

    // ======================= FP3 ===========================================
    k_fill<<<1, 256, 0, stream>>>(zero3, 3);
    k_knn_small<<<g1d(82), 256, 0, stream>>>(cpos2, 82, zero3, 1, 1, iI3, wI3);
    k_fill<<<g1d(96ll * 3072), 256, 0, stream>>>(slotA, 96ll * 3072);
    k_interp<<<g1d(82ll * 2048), 256, 0, stream>>>(x3, 2048, iI3, wI3, 1, 82, 2048, slotA, 3072);
    k_copy2d<<<g1d(82ll * 1024), 256, 0, stream>>>(x2, 1024, 82, 1024, slotA, 3072, 2048);
    mlpLayer(9,  slotA, 3072, 82, 96, nullptr, nullptr, slotB);    // -> [*,1024]
    mlpLayer(10, slotB, 1024, 82, 96, nullptr, nullptr, slotA);    // g3 = slotA [82,1024]

    // ======================= FP2 ===========================================
    k_knn_small<<<g1d(1638), 256, 0, stream>>>(cpos1, 1638, cpos2, 82, 3, iI2, wI2);
    k_fill<<<g1d(1648ll * 1536), 256, 0, stream>>>(slotB, 1648ll * 1536);
    k_interp<<<g1d(1638ll * 1024), 256, 0, stream>>>(slotA, 1024, iI2, wI2, 3, 1638, 1024, slotB, 1536);
    k_copy2d<<<g1d(1638ll * 512), 256, 0, stream>>>(x1, 512, 1638, 512, slotB, 1536, 1024);
    mlpLayer(11, slotB, 1536, 1638, 1648, nullptr, nullptr, slotA);
    mlpLayer(12, slotA, 1024, 1638, 1648, nullptr, nullptr, slotB); // y2 = slotB

    // ======================= FP1 ===========================================
    k_knn_small<<<g1d(16384), 256, 0, stream>>>(pos, 16384, cpos1, 1638, 3, iI1, wI1);
    k_fill<<<g1d(16384ll * 1056), 256, 0, stream>>>(slotA, 16384ll * 1056);
    k_interp<<<g1d(16384ll * 1024), 256, 0, stream>>>(slotB, 1024, iI1, wI1, 3, 16384, 1024, slotA, 1056);
    k_copy2d<<<g1d(16384ll * 3), 256, 0, stream>>>(xin, 3, 16384, 3, slotA, 1056, 1024);
    mlpLayer(13, slotA, 1056, 16384, 16384, nullptr, nullptr, slotB);
    mlpLayer(14, slotB, 1024, 16384, 16384, nullptr, nullptr, slotA); // y1 = slotA

    // ======================= Head ==========================================
    gemm(slotA, 1024, 16384, wd[15], P(63), 0, slotB);                 // no relu before BN
    bn(slotB, 16384, 16384, 1024, nullptr, nullptr, P(64), P(65), 1);  // relu after BN
    gemm(slotB, 1024, 16384, wd[16], P(67), 0, lg);                    // logits [16384, 128(13)]
    k_lsm<<<g1d(16384), 256, 0, stream>>>(lg, 128, 16384, (float*)d_out);
}
